// HyperbolicLogisticLayer_47476568490613
// MI455X (gfx1250) — matrix operand order repaired and checked by reading, not compiled, not run on hardware
//
#include <hip/hip_runtime.h>
#include <math.h>

// ---------------------------------------------------------------------------
// HyperbolicLogisticLayer on MI455X (gfx1250, wave32, WMMA).
// B=4096, U=256, D=128. Two shared-A GEMMs (x·W^T, x·bias^T) via
// v_wmma_f32_16x16x32_bf16, fused elementwise hyperbolic math + softmax.
// ---------------------------------------------------------------------------

typedef __attribute__((ext_vector_type(16))) __bf16 v16bf;
typedef __attribute__((ext_vector_type(8)))  float  v8f;

#define B_DIM 4096
#define U_DIM 256
#define D_DIM 128
#define B_TILE 16

static __device__ __forceinline__ v8f wmma_bf16(v16bf a, v16bf b, v8f c) {
  // 8 args: (neg_a, A, neg_b, B, c_mod, C, reuse_a, reuse_b)
  return __builtin_amdgcn_wmma_f32_16x16x32_bf16(
      false, a, false, b, (short)0, c, false, false);
}

// ---------------------------------------------------------------------------
// Kernel 1: per-u scalars.
// scal[0*256+u]=nb2, [1]=1-nb2, [2]=dot(bias,W), [3]=(1-nb2)*||W||, [4]=2*||W||
// ---------------------------------------------------------------------------
__global__ void __launch_bounds__(256)
prep_scalars_kernel(const float* __restrict__ W, const float* __restrict__ bias,
                    float* __restrict__ scal) {
  const int u = threadIdx.x;
  const float* wr = W + u * D_DIM;
  const float* br = bias + u * D_DIM;
  float nb2 = 0.f, bw = 0.f, wn2 = 0.f;
#pragma unroll 4
  for (int k = 0; k < D_DIM; ++k) {
    const float w = wr[k], b = br[k];
    nb2 = fmaf(b, b, nb2);
    bw  = fmaf(b, w, bw);
    wn2 = fmaf(w, w, wn2);
  }
  const float omn = 1.0f - nb2;
  const float swn = sqrtf(wn2);
  scal[0 * U_DIM + u] = nb2;
  scal[1 * U_DIM + u] = omn;
  scal[2 * U_DIM + u] = bw;
  scal[3 * U_DIM + u] = omn * swn;   // a_norm
  scal[4 * U_DIM + u] = 2.0f * swn;  // lam * a_norm
}

// ---------------------------------------------------------------------------
// Kernel 2: pack W and bias (row-major U x D f32) into bf16 WMMA B-fragment
// order. For B (32x16, KxN) in this file's contiguous k order (same as A): lane l holds column N = l&15,
// K range = 16*(l>>4) + j (j = 0..15 = VGPR pair index). Packed so each lane
// reads its 32 contiguous bytes: dst[((ut*4+kc)*32 + l)*16 + j].
// ---------------------------------------------------------------------------
__global__ void __launch_bounds__(256)
pack_bf16_kernel(const float* __restrict__ W, const float* __restrict__ bias,
                 __bf16* __restrict__ Wpk, __bf16* __restrict__ Bpk) {
  const int idx = blockIdx.x * 256 + threadIdx.x;   // 0 .. 32767
  const int j  = idx & 15;
  const int l  = (idx >> 4) & 31;
  const int kc = (idx >> 9) & 3;
  const int ut = idx >> 11;
  const int u  = ut * 16 + (l & 15);
  const int k  = kc * 32 + (l >> 4) * 16 + j;
  Wpk[idx] = (__bf16)W[u * D_DIM + k];
  Bpk[idx] = (__bf16)bias[u * D_DIM + k];
}

// ---------------------------------------------------------------------------
// Kernel 3: main fused kernel. One block = 16 rows of x, all 256 u.
// 8 wave32 per block; wave w handles u-tiles {2w, 2w+1}.
// ---------------------------------------------------------------------------
__global__ void __launch_bounds__(256)
hyp_softmax_kernel(const float* __restrict__ x,
                   const __bf16* __restrict__ Wpk,
                   const __bf16* __restrict__ Bpk,
                   const float* __restrict__ scal,
                   float* __restrict__ out) {
  __shared__ float s_logits[B_TILE][U_DIM];   // 16 KB
  __shared__ float s_red[B_TILE][16];
  __shared__ float s_nv2[B_TILE];
  __shared__ float s_rowmax[B_TILE];
  __shared__ float s_rowsum[B_TILE];
  __shared__ float s_nb2[U_DIM], s_omn[U_DIM], s_bw[U_DIM], s_an[U_DIM], s_sc[U_DIM];

  const int tid  = threadIdx.x;
  const int lane = tid & 31;
  const int wave = tid >> 5;
  const int btile = blockIdx.x;

  // stage per-u scalars into LDS
  s_nb2[tid] = scal[0 * U_DIM + tid];
  s_omn[tid] = scal[1 * U_DIM + tid];
  s_bw[tid]  = scal[2 * U_DIM + tid];
  s_an[tid]  = scal[3 * U_DIM + tid];
  s_sc[tid]  = scal[4 * U_DIM + tid];

  // nv2 = ||x_b||^2 per row (16 segs of 8 elements each)
  {
    const int row = tid >> 4, seg = tid & 15;
    const float* xr = x + (size_t)(btile * B_TILE + row) * D_DIM + seg * 8;
    float s = 0.f;
#pragma unroll
    for (int i = 0; i < 8; ++i) s = fmaf(xr[i], xr[i], s);
    s_red[row][seg] = s;
  }
  __syncthreads();
  if (tid < B_TILE) {
    float s = 0.f;
#pragma unroll
    for (int i = 0; i < 16; ++i) s += s_red[tid][i];
    s_nv2[tid] = s;
  }
  __syncthreads();

  // Load A fragments (x tile, 16x128) in contiguous-k 16-bit A layout (matches packed B):
  // lane l holds row M=l&15; element j -> K = j + 16*(l>>4) + 32*c
  v16bf afrag[4];
  {
    const float* xrow = x + (size_t)(btile * B_TILE + (lane & 15)) * D_DIM;
    const int hl = lane >> 4;
#pragma unroll
    for (int c = 0; c < 4; ++c) {
      const int base = 32 * c + 16 * hl;
      const float4 f0 = *(const float4*)(xrow + base);
      const float4 f1 = *(const float4*)(xrow + base + 4);
      const float4 f2 = *(const float4*)(xrow + base + 8);
      const float4 f3 = *(const float4*)(xrow + base + 12);
      const float tmp[16] = {f0.x, f0.y, f0.z, f0.w, f1.x, f1.y, f1.z, f1.w,
                             f2.x, f2.y, f2.z, f2.w, f3.x, f3.y, f3.z, f3.w};
#pragma unroll
      for (int j = 0; j < 16; ++j) afrag[c][j] = (__bf16)tmp[j];
    }
  }

  // Two u-tiles per wave: GEMMs + fused hyperbolic elementwise
#pragma unroll
  for (int t = 0; t < 2; ++t) {
    const int ut = wave * 2 + t;
    v8f accW = {};  // S1 = x . W
    v8f accB = {};  // S2 = x . bias
#pragma unroll
    for (int c = 0; c < 4; ++c) {
      const size_t boff = (size_t)((ut * 4 + c) * 32 + lane) * 16;
      const v16bf bwf = *(const v16bf*)(Wpk + boff);
      const v16bf bbf = *(const v16bf*)(Bpk + boff);
      accW = wmma_bf16(afrag[c], bwf, accW);
      accB = wmma_bf16(afrag[c], bbf, accB);
    }
    const int u = ut * 16 + (lane & 15);
    const float nb2 = s_nb2[u], omn = s_omn[u], bw = s_bw[u];
    const float an = s_an[u], sc = s_sc[u];
#pragma unroll
    for (int r = 0; r < 8; ++r) {
      const int   m   = r + 8 * (lane >> 4);
      const float S1  = accW[r];
      const float S2  = accB[r];
      const float nv2 = s_nv2[m];
      const float dd    = fmaf(nb2, nv2, 1.0f - 2.0f * S2);
      const float rinv  = 1.0f / dd;
      const float alpha = (1.0f - 2.0f * S2 + nv2) * rinv;
      const float beta  = omn * rinv;
      const float num   = 2.0f * omn * (beta * S1 - alpha * bw);
      const float mm    = alpha * alpha * nb2 - 2.0f * alpha * beta * S2 + beta * beta * nv2;
      const float den   = (1.0f - mm) * an;
      const float tt    = num / den;
      const float att   = fabsf(tt);
      const float asnh  = logf(att + sqrtf(fmaf(att, att, 1.0f)));
      s_logits[m][u] = sc * copysignf(asnh, tt);
    }
  }
  __syncthreads();

  // Fused softmax over U=256 per row (16 threads per row, 16 values each)
  const int row = tid >> 4, seg = tid & 15;
  {
    float mx = -3.0e38f;
#pragma unroll
    for (int j = 0; j < 16; ++j) mx = fmaxf(mx, s_logits[row][seg * 16 + j]);
    s_red[row][seg] = mx;
  }
  __syncthreads();
  if (tid < B_TILE) {
    float mx = -3.0e38f;
#pragma unroll
    for (int j = 0; j < 16; ++j) mx = fmaxf(mx, s_red[tid][j]);
    s_rowmax[tid] = mx;
  }
  __syncthreads();
  {
    const float rm = s_rowmax[row];
    float sum = 0.f;
#pragma unroll
    for (int j = 0; j < 16; ++j) {
      const float e = __expf(s_logits[row][seg * 16 + j] - rm);
      s_logits[row][seg * 16 + j] = e;
      sum += e;
    }
    s_red[row][seg] = sum;
  }
  __syncthreads();
  if (tid < B_TILE) {
    float s = 0.f;
#pragma unroll
    for (int j = 0; j < 16; ++j) s += s_red[tid][j];
    s_rowsum[tid] = s;
  }
  __syncthreads();
  {
    const float inv = 1.0f / s_rowsum[row];
    float* orow = out + (size_t)(btile * B_TILE + row) * U_DIM + seg * 16;
#pragma unroll
    for (int j = 0; j < 16; ++j) orow[j] = s_logits[row][seg * 16 + j] * inv;
  }
}

// ---------------------------------------------------------------------------
extern "C" void kernel_launch(void* const* d_in, const int* in_sizes, int n_in,
                              void* d_out, int out_size, void* d_ws, size_t ws_size,
                              hipStream_t stream) {
  const float* x    = (const float*)d_in[0];  // 4096 x 128
  const float* W    = (const float*)d_in[1];  // 256 x 128
  const float* bias = (const float*)d_in[2];  // 256 x 128
  float* out = (float*)d_out;                 // 4096 x 256

  // workspace: Wpk bf16 (64KB) | Bpk bf16 (64KB) | scalars 5*256 f32 (5KB)
  __bf16* Wpk = (__bf16*)d_ws;
  __bf16* Bpk = Wpk + (U_DIM * D_DIM);
  float*  scal = (float*)((char*)d_ws + 2 * U_DIM * D_DIM * sizeof(__bf16));

  prep_scalars_kernel<<<1, 256, 0, stream>>>(W, bias, scal);
  pack_bf16_kernel<<<(U_DIM * D_DIM) / 256, 256, 0, stream>>>(W, bias, Wpk, Bpk);
  hyp_softmax_kernel<<<B_DIM / B_TILE, 256, 0, stream>>>(x, Wpk, Bpk, scal, out);
}
